// BGHGNN_82952998355820
// MI455X (gfx1250) — matrix operand order repaired and checked by reading, not compiled, not run on hardware
//
#include <hip/hip_runtime.h>
#include <hip/hip_bf16.h>
#include <math.h>

typedef __attribute__((ext_vector_type(16))) __bf16 v16bf;
typedef __attribute__((ext_vector_type(8)))  float  v8f;

#define NCOLS 256   // all GEMMs in this net have 256 output columns
#define HEADS 4
#define DDIM  64

// ===========================================================================
// Pack kernels: f32 -> bf16 in WMMA fragment-native layout, K padded to 32.
//
// A fragment (ISA 7.12.2, 16-bit A 16x32): lane = m + 16*half,
//   element i (0..15) -> K = half*16 + i (contiguous like B, not the ISA table order)
// B fragment (32x16): lane = col + 16*half, element i -> K = half*16 + i
// Packed layout: [tile][chunk][lane][elem] contiguous -> each fragment is one
// 32-byte contiguous per-lane load in the GEMM (2x global_load_b128).
// ===========================================================================
__global__ void __launch_bounds__(256)
pack_A(const float* __restrict__ A, __bf16* __restrict__ Ap,
       int M, int K, int KC) {
  size_t idx = (size_t)blockIdx.x * blockDim.x + threadIdx.x;
  size_t total = (size_t)((M + 15) / 16) * KC * 512;
  if (idx >= total) return;
  int i     = (int)(idx & 15);
  int lane  = (int)((idx >> 4) & 31);
  size_t cc = idx >> 9;               // tile*KC + chunk
  int chunk = (int)(cc % KC);
  int tile  = (int)(cc / KC);
  int m = lane & 15, half = lane >> 4;
  int kk = half * 16 + i;
  int n = tile * 16 + m, k = chunk * 32 + kk;
  float v = (n < M && k < K) ? A[(size_t)n * K + k] : 0.0f;
  Ap[idx] = (__bf16)v;
}

// blocked==1: B is (R,K,64) row-major viewed as (K,256), col j=(r*64+c)
__global__ void __launch_bounds__(256)
pack_B(const float* __restrict__ B, __bf16* __restrict__ Bp,
       int K, int KC, int blocked) {
  size_t idx = (size_t)blockIdx.x * blockDim.x + threadIdx.x;
  size_t total = (size_t)16 * KC * 512;   // 16 col-tiles
  if (idx >= total) return;
  int i     = (int)(idx & 15);
  int lane  = (int)((idx >> 4) & 31);
  size_t cc = idx >> 9;               // ct*KC + chunk
  int chunk = (int)(cc % KC);
  int ct    = (int)(cc / KC);
  int col = ct * 16 + (lane & 15);
  int k = chunk * 32 + (lane >> 4) * 16 + i;
  float v = 0.0f;
  if (k < K)
    v = blocked ? B[((size_t)(col >> 6) * K + k) * 64 + (col & 63)]
                : B[(size_t)k * NCOLS + col];
  Bp[idx] = (__bf16)v;
}

// ===========================================================================
// WMMA GEMM on packed bf16: C(M x 256) = A(M x Kp) * B(Kp x 256)
// Block: one 16-row tile, 128 threads (4 waves); wave w owns cols [64w,64w+64)
// Inner loop: 1 A-frag load + 4 B-frag loads + 4 wmma, branch-free.
// ===========================================================================
__global__ void __launch_bounds__(128)
gemm_packed_wmma(const __bf16* __restrict__ Ap, const __bf16* __restrict__ Bp,
                 float* __restrict__ C, int M, int KC) {
  const int tid  = threadIdx.x;
  const int lane = tid & 31;
  const int wave = tid >> 5;
  const int half = lane >> 4;
  const int l16  = lane & 15;
  const int tile = blockIdx.x;

  const v16bf* __restrict__ Av = (const v16bf*)Ap;
  const v16bf* __restrict__ Bv = (const v16bf*)Bp;

  v8f acc[4] = {v8f{}, v8f{}, v8f{}, v8f{}};

  for (int c = 0; c < KC; ++c) {
    // speculative prefetch of next A chunk (global_prefetch_b8)
    __builtin_prefetch(&Av[((size_t)tile * KC + c + 1) * 32 + lane], 0, 1);

    v16bf a = Av[((size_t)tile * KC + c) * 32 + lane];
#pragma unroll
    for (int t = 0; t < 4; ++t) {
      const int ct = wave * 4 + t;
      v16bf b = Bv[((size_t)ct * KC + c) * 32 + lane];
      acc[t] = __builtin_amdgcn_wmma_f32_16x16x32_bf16(
          false, a, false, b, (short)0, acc[t], false, false);
    }
  }

  // C/D layout: element jj -> (M = jj + 8*half, N = l16)
  const int row0 = tile * 16;
#pragma unroll
  for (int t = 0; t < 4; ++t) {
    const int cix = wave * 64 + t * 16 + l16;
#pragma unroll
    for (int jj = 0; jj < 8; ++jj) {
      const int rr = row0 + jj + half * 8;
      if (rr < M) C[(size_t)rr * NCOLS + cix] = acc[t][jj];
    }
  }
}

// ===========================================================================
// Elementwise / edge kernels
// ===========================================================================
__global__ void combine_elu(const float* __restrict__ fc,
                            const float* __restrict__ ft,
                            const float* __restrict__ fr,
                            const float* __restrict__ fw,
                            const float* __restrict__ fb,
                            float* __restrict__ h0, int Mn) {
  int idx = blockIdx.x * blockDim.x + threadIdx.x;  // n*64 + c
  if (idx >= Mn * 64) return;
  int n = idx >> 6, c = idx & 63;
  float acc = 0.0f;
#pragma unroll
  for (int rr = 0; rr < 4; ++rr) {
    size_t o = (size_t)n * 256 + rr * 64 + c;
    acc += fw[rr] * fc[o] * ft[o] * fr[o];
  }
  acc += fb[c];
  h0[idx] = acc > 0.0f ? acc : (expf(acc) - 1.0f);
}

__global__ void compute_elr(const float* __restrict__ feat,
                            const float* __restrict__ al,
                            const float* __restrict__ ar,
                            float* __restrict__ el, float* __restrict__ er,
                            int Mn) {
  int idx = blockIdx.x * blockDim.x + threadIdx.x;  // n*4 + h
  if (idx >= Mn * HEADS) return;
  int n = idx >> 2, h = idx & 3;
  const float* f = feat + (size_t)n * NCOLS + h * DDIM;
  float sl = 0.0f, sr = 0.0f;
#pragma unroll 4
  for (int d = 0; d < DDIM; ++d) {
    float v = f[d];
    sl += v * al[h * DDIM + d];
    sr += v * ar[h * DDIM + d];
  }
  el[idx] = sl;
  er[idx] = sr;
}

// ordered-uint encoding for atomic float max (handles negatives)
__device__ __forceinline__ unsigned encf(float f) {
  unsigned u = __float_as_uint(f);
  return (u & 0x80000000u) ? ~u : (u | 0x80000000u);
}
__device__ __forceinline__ float decf(unsigned u) {
  return __uint_as_float((u & 0x80000000u) ? (u & 0x7FFFFFFFu) : ~u);
}

__global__ void init_layer(unsigned* __restrict__ m, float* __restrict__ s,
                           float* __restrict__ agg, int Mn) {
  int i = blockIdx.x * blockDim.x + threadIdx.x;
  if (i < Mn * HEADS) { m[i] = encf(-INFINITY); s[i] = 0.0f; }
  if (i < Mn * NCOLS) agg[i] = 0.0f;
}

__global__ void edge_max(const int* __restrict__ src, const int* __restrict__ dst,
                         const float* __restrict__ el, const float* __restrict__ er,
                         float* __restrict__ ebuf, unsigned* __restrict__ m,
                         int Ecnt) {
  int idx = blockIdx.x * blockDim.x + threadIdx.x;  // e*4 + h
  if (idx >= Ecnt * HEADS) return;
  int e = idx >> 2, h = idx & 3;
  float x = el[src[e] * HEADS + h] + er[dst[e] * HEADS + h];
  x = x > 0.0f ? x : 0.2f * x;
  ebuf[idx] = x;
  atomicMax(&m[dst[e] * HEADS + h], encf(x));
}

__global__ void edge_exp(const int* __restrict__ dst, float* __restrict__ ebuf,
                         const unsigned* __restrict__ m, float* __restrict__ s,
                         int Ecnt) {
  int idx = blockIdx.x * blockDim.x + threadIdx.x;
  if (idx >= Ecnt * HEADS) return;
  int e = idx >> 2, h = idx & 3;
  float ee = expf(ebuf[idx] - decf(m[dst[e] * HEADS + h]));
  ebuf[idx] = ee;
  atomicAdd(&s[dst[e] * HEADS + h], ee);
}

__global__ void __launch_bounds__(256)
edge_msg(const int* __restrict__ src, const int* __restrict__ dst,
         const float* __restrict__ ebuf, const float* __restrict__ s,
         const float* __restrict__ feat, float* __restrict__ agg,
         float* __restrict__ att, int Ecnt) {
  int e = blockIdx.x;
  if (e >= Ecnt) return;
  int hd = threadIdx.x;           // h*64 + d
  int h = hd >> 6;
  int sn = src[e], dn = dst[e];
  float a = ebuf[e * HEADS + h] / s[dn * HEADS + h];
  if (att && hd < HEADS) att[e * HEADS + hd] = ebuf[e * HEADS + hd] / s[dn * HEADS + hd];
  atomicAdd(&agg[(size_t)dn * NCOLS + hd], a * feat[(size_t)sn * NCOLS + hd]);
}

__global__ void finalize_out(const float* __restrict__ agg,
                             const float* __restrict__ bias,
                             const float* __restrict__ resid,
                             float* __restrict__ out, int Mn) {
  int i = blockIdx.x * blockDim.x + threadIdx.x;
  if (i >= Mn * NCOLS) return;
  float v = agg[i] + bias[i & (NCOLS - 1)];
  if (resid) v += resid[i];
  out[i] = v;
}

// ===========================================================================
// host-side orchestration
// ===========================================================================
extern "C" void kernel_launch(void* const* d_in, const int* in_sizes, int n_in,
                              void* d_out, int out_size, void* d_ws, size_t ws_size,
                              hipStream_t stream) {
  (void)n_in; (void)out_size; (void)ws_size;
  const int*   src  = (const int*)d_in[0];
  const int*   dst  = (const int*)d_in[1];
  const float* hc   = (const float*)d_in[2];   // (N,129)
  const float* ht   = (const float*)d_in[3];   // (N,513)
  const float* hr   = (const float*)d_in[4];   // (N,513)
  const float* Wc   = (const float*)d_in[5];   // (4,129,64)
  const float* Wt   = (const float*)d_in[6];   // (4,513,64)
  const float* Wr   = (const float*)d_in[7];   // (4,513,64)
  const float* fw   = (const float*)d_in[8];   // (1,4)
  const float* fb   = (const float*)d_in[9];   // (1,64)
  const float* W0   = (const float*)d_in[10];  // (64,256)
  const float* al0  = (const float*)d_in[11];
  const float* ar0  = (const float*)d_in[12];
  const float* b0   = (const float*)d_in[13];
  const float* W1   = (const float*)d_in[14];  // (256,256)
  const float* al1  = (const float*)d_in[15];
  const float* ar1  = (const float*)d_in[16];
  const float* b1   = (const float*)d_in[17];

  const int E = in_sizes[0];
  const int N = in_sizes[2] / 129;
  const int tiles = (N + 15) / 16;
  const int KCc = 5, KCt = 17, KC0 = 2, KC1 = 8;  // ceil(K/32)

  // -------- workspace layout (floats, with reuse) --------
  float* bufA = (float*)d_ws;                       // N*256: fcA -> feat
  float* bufB = bufA + (size_t)N * 256;             // N*256: ftA -> agg
  float* bufC = bufB + (size_t)N * 256;             // N*256: frA -> h1
  float* h0   = bufC + (size_t)N * 256;             // N*64
  float* ebuf = h0   + (size_t)N * 64;              // E*4
  unsigned* menc = (unsigned*)(ebuf + (size_t)E * 4); // N*4
  float* sden = (float*)(menc + (size_t)N * 4);     // N*4
  float* el   = sden + (size_t)N * 4;               // N*4
  float* er   = el   + (size_t)N * 4;               // N*4
  // bf16 pack regions
  __bf16* Apack = (__bf16*)(er + (size_t)N * 4);    // tiles*17*512 (max KC)
  __bf16* Bc = Apack + (size_t)tiles * KCt * 512;   // 16*KCc*512
  __bf16* Bt = Bc + (size_t)16 * KCc * 512;
  __bf16* Br = Bt + (size_t)16 * KCt * 512;
  __bf16* B0 = Br + (size_t)16 * KCt * 512;
  __bf16* B1 = B0 + (size_t)16 * KC0 * 512;

  float* hout   = (float*)d_out;                    // (N,256)
  float* attout = hout + (size_t)N * 256;           // (E,4)

  const int thr = 256;
  dim3 blk128(128), blk256(256);
  auto nblk = [](size_t n, int t) { return (int)((n + t - 1) / t); };
  const size_t apTot5  = (size_t)tiles * KCc * 512;
  const size_t apTot17 = (size_t)tiles * KCt * 512;
  const size_t apTot2  = (size_t)tiles * KC0 * 512;
  const size_t apTot8  = (size_t)tiles * KC1 * 512;

  // -------- pack all weights (tiny, L2-resident afterwards) --------
  pack_B<<<nblk((size_t)16 * KCc * 512, thr), blk256, 0, stream>>>(Wc, Bc, 129, KCc, 1);
  pack_B<<<nblk((size_t)16 * KCt * 512, thr), blk256, 0, stream>>>(Wt, Bt, 513, KCt, 1);
  pack_B<<<nblk((size_t)16 * KCt * 512, thr), blk256, 0, stream>>>(Wr, Br, 513, KCt, 1);
  pack_B<<<nblk((size_t)16 * KC0 * 512, thr), blk256, 0, stream>>>(W0, B0, 64, KC0, 0);
  pack_B<<<nblk((size_t)16 * KC1 * 512, thr), blk256, 0, stream>>>(W1, B1, 256, KC1, 0);

  // -------- trilinear fusion: three GEMMs, then combine+ELU --------
  pack_A<<<nblk(apTot5, thr), blk256, 0, stream>>>(hc, Apack, N, 129, KCc);
  gemm_packed_wmma<<<tiles, blk128, 0, stream>>>(Apack, Bc, bufA, N, KCc);
  pack_A<<<nblk(apTot17, thr), blk256, 0, stream>>>(ht, Apack, N, 513, KCt);
  gemm_packed_wmma<<<tiles, blk128, 0, stream>>>(Apack, Bt, bufB, N, KCt);
  pack_A<<<nblk(apTot17, thr), blk256, 0, stream>>>(hr, Apack, N, 513, KCt);
  gemm_packed_wmma<<<tiles, blk128, 0, stream>>>(Apack, Br, bufC, N, KCt);
  combine_elu<<<nblk((size_t)N * 64, thr), blk256, 0, stream>>>(bufA, bufB, bufC,
                                                                fw, fb, h0, N);

  // -------- GAT layer 0: h0(N,64) -> h1(N,256) --------
  pack_A<<<nblk(apTot2, thr), blk256, 0, stream>>>(h0, Apack, N, 64, KC0);
  gemm_packed_wmma<<<tiles, blk128, 0, stream>>>(Apack, B0, bufA, N, KC0);
  compute_elr<<<nblk((size_t)N * 4, thr), blk256, 0, stream>>>(bufA, al0, ar0, el, er, N);
  init_layer<<<nblk((size_t)N * 256, thr), blk256, 0, stream>>>(menc, sden, bufB, N);
  edge_max<<<nblk((size_t)E * 4, thr), blk256, 0, stream>>>(src, dst, el, er, ebuf, menc, E);
  edge_exp<<<nblk((size_t)E * 4, thr), blk256, 0, stream>>>(dst, ebuf, menc, sden, E);
  edge_msg<<<E, blk256, 0, stream>>>(src, dst, ebuf, sden, bufA, bufB, (float*)nullptr, E);
  finalize_out<<<nblk((size_t)N * 256, thr), blk256, 0, stream>>>(bufB, b0,
                                                                  (const float*)nullptr,
                                                                  bufC, N);

  // -------- GAT layer 1: h1(N,256) -> out(N,256), residual, attention --------
  pack_A<<<nblk(apTot8, thr), blk256, 0, stream>>>(bufC, Apack, N, 256, KC1);
  gemm_packed_wmma<<<tiles, blk128, 0, stream>>>(Apack, B1, bufA, N, KC1);
  compute_elr<<<nblk((size_t)N * 4, thr), blk256, 0, stream>>>(bufA, al1, ar1, el, er, N);
  init_layer<<<nblk((size_t)N * 256, thr), blk256, 0, stream>>>(menc, sden, bufB, N);
  edge_max<<<nblk((size_t)E * 4, thr), blk256, 0, stream>>>(src, dst, el, er, ebuf, menc, E);
  edge_exp<<<nblk((size_t)E * 4, thr), blk256, 0, stream>>>(dst, ebuf, menc, sden, E);
  edge_msg<<<E, blk256, 0, stream>>>(src, dst, ebuf, sden, bufA, bufB, attout, E);
  finalize_out<<<nblk((size_t)N * 256, thr), blk256, 0, stream>>>(bufB, b1, bufC, hout, N);
}
